// Sequence_10161892622885
// MI455X (gfx1250) — matrix operand order repaired and checked by reading, not compiled, not run on hardware
//
#include <hip/hip_runtime.h>

typedef __attribute__((ext_vector_type(16))) _Float16 v16h;
typedef __attribute__((ext_vector_type(8)))  float    v8f;

#define HH   51
#define G4   204          // 4*H gate columns
#define NT   13           // N tiles (208 padded columns)
#define GP   212          // padded gate row stride (bank-conflict-free)
#define WPB  4            // waves per block
#define RPW  16           // batch rows per wave
#define TLEN 1024

// ---- fast transcendentals: prefer hardware v_tanh_f32 on gfx1250 ----
#if __has_builtin(__builtin_amdgcn_tanhf)
__device__ __forceinline__ float tanh_f(float x) {
    return __builtin_amdgcn_tanhf(x);
}
__device__ __forceinline__ float sig_f(float x) {
    return __builtin_fmaf(0.5f, __builtin_amdgcn_tanhf(0.5f * x), 0.5f);
}
#elif __has_builtin(__builtin_amdgcn_tanh_f32)
__device__ __forceinline__ float tanh_f(float x) {
    return __builtin_amdgcn_tanh_f32(x);
}
__device__ __forceinline__ float sig_f(float x) {
    return __builtin_fmaf(0.5f, __builtin_amdgcn_tanh_f32(0.5f * x), 0.5f);
}
#else
__device__ __forceinline__ float sig_f(float x) {
    return __builtin_amdgcn_rcpf(1.0f + __builtin_amdgcn_exp2f(x * -1.44269504088896340736f));
}
__device__ __forceinline__ float tanh_f(float x) {
    return __builtin_fmaf(2.0f, sig_f(2.0f * x), -1.0f);
}
#endif

// ---------------------------------------------------------------------------
// Prep: pack weights into WMMA B-matrix f16 images.
// Bimg layout: [layer][tile][chunk][lane 0..31][slot 0..15]
//   lane l holds column n = 16*tile + (l&15), K = chunk*32 + (slot&7) + 8*(l>>4) + 16*(slot>>3)
// Synthetic K rows (bias/input folded into the GEMM):
//   k = 51 : fused bias  (b_ih + b_hh)          -- pairs with A row == 1.0
//   k = 52 : W_ih0 column (layer 0) / 0 (layer1)-- pairs with A row == x_t[m]
// fbuf: wl[64]
// ---------------------------------------------------------------------------
__global__ void lstm_prep_kernel(const float* __restrict__ W_ih0,
                                 const float* __restrict__ W_hh0,
                                 const float* __restrict__ b_ih0,
                                 const float* __restrict__ b_hh0,
                                 const float* __restrict__ W_ih1,
                                 const float* __restrict__ W_hh1,
                                 const float* __restrict__ b_ih1,
                                 const float* __restrict__ b_hh1,
                                 const float* __restrict__ Wl,
                                 _Float16* __restrict__ Bimg,
                                 float* __restrict__ fbuf) {
    const int tid = threadIdx.x;
    const int total = 2 * NT * 2 * 32 * 16;
    for (int idx = tid; idx < total; idx += blockDim.x) {
        int s     = idx & 15;
        int l     = (idx >> 4) & 31;
        int ch    = (idx >> 9) & 1;
        int tt    = (idx >> 10) % NT;
        int layer = (idx >> 10) / NT;
        int n = 16 * tt + (l & 15);
        int k = ch * 32 + (s & 7) + 8 * (l >> 4) + 16 * (s >> 3);
        float v = 0.0f;
        if (n < G4) {
            if (k < HH) {
                v = (layer == 0) ? W_hh0[n * HH + k]
                                 : (W_ih1[n * HH + k] + W_hh1[n * HH + k]);
            } else if (k == HH) {        // bias row
                v = (layer == 0) ? (b_ih0[n] + b_hh0[n]) : (b_ih1[n] + b_hh1[n]);
            } else if (k == HH + 1) {    // input-projection row (layer 0 only)
                v = (layer == 0) ? W_ih0[n] : 0.0f;
            }
        }
        Bimg[idx] = (_Float16)v;
    }
    for (int n = tid; n < 64; n += blockDim.x)
        fbuf[n] = (n < HH) ? Wl[n] : 0.0f;
}

// ---------------------------------------------------------------------------
// Main persistent-per-wave LSTM scan. One wave owns 16 batch rows for all T.
// ---------------------------------------------------------------------------
__global__ __launch_bounds__(WPB * 32) void lstm_scan_kernel(
        const float* __restrict__ input,      // (B, T)
        const _Float16* __restrict__ Bimg,
        const float* __restrict__ fbuf,
        const float* __restrict__ blp,        // bl scalar
        float* __restrict__ out,              // (B, T)
        int B) {
    __shared__ __align__(32) _Float16 sB[2][NT][2][32][16];   // 52 KB weights
    __shared__ __align__(32) _Float16 hA[WPB][2][32][16];     // A-image of [h,1,x]
    __shared__ float gates[WPB][16][GP];                      // 54 KB scratch
    __shared__ float sWl[64];
    __shared__ float obuf[WPB][32];

    const int tid  = threadIdx.x;
    const int w    = tid >> 5;
    const int lane = tid & 31;

    // cooperative weight load into LDS
    {
        const uint4* src = (const uint4*)Bimg;
        uint4* dst = (uint4*)&sB[0][0][0][0][0];
        for (int i = tid; i < (2 * NT * 2 * 32 * 16 * 2) / 16; i += blockDim.x)
            dst[i] = src[i];
        for (int i = tid; i < 64; i += blockDim.x) sWl[i] = fbuf[i];
    }
    // zero h image (initial h = 0; permanent K padding), then set A row j=51 = 1.0
    for (int i = lane; i < 2 * 32 * 16; i += 32)
        (&hA[w][0][0][0])[i] = (_Float16)0.0f;
    if (lane < 16)
        hA[w][1][lane][11] = (_Float16)1.0f;   // j=51 -> chunk1, lane m, slot 11
    __syncthreads();

    const int hi    = lane >> 4;       // half-wave id
    const int lo    = lane & 15;
    const int m     = lo;              // owned batch row (elementwise phase)
    const int jbase = hi * 26;         // owned hidden range start
    const int rowBase = (blockIdx.x * WPB + w) * RPW;
    const float bl = blp[0];

    float c[26];
    #pragma unroll
    for (int k = 0; k < 26; ++k) c[k] = 0.0f;

    // software-pipelined x load
    float xv = 0.0f;
    if (lane < 16) xv = input[(size_t)(rowBase + lane) * TLEN];

    for (int t = 0; t < TLEN; ++t) {
        // ---- publish x_t into A row j=52 (chunk1, lane m, slot 12) ----
        if (lane < 16)
            hA[w][1][lane][12] = (_Float16)xv;
        asm volatile("" ::: "memory");
        // prefetch x_{t+1} (latency hidden behind the step's compute)
        float xnext = 0.0f;
        if (lane < 16 && t + 1 < TLEN)
            xnext = input[(size_t)(rowBase + lane) * TLEN + t + 1];

        // ---- layer 0: gates = [h,1,x] @ [Whh0; b0; Wih0]^T ----
        v16h a0 = *(const v16h*)&hA[w][0][lane][0];
        v16h a1 = *(const v16h*)&hA[w][1][lane][0];
        for (int nt = 0; nt < NT; ++nt) {
            const int col = nt * 16 + lo;
            v8f acc = {};
            v16h b0v = *(const v16h*)&sB[0][nt][0][lane][0];
            v16h b1v = *(const v16h*)&sB[0][nt][1][lane][0];
            acc = __builtin_amdgcn_wmma_f32_16x16x32_f16(false, a0, false, b0v,
                                                         (short)0, acc, false, false);
            acc = __builtin_amdgcn_wmma_f32_16x16x32_f16(false, a1, false, b1v,
                                                         (short)0, acc, false, false);
            #pragma unroll
            for (int r = 0; r < 8; ++r) gates[w][hi * 8 + r][col] = acc[r];
        }
        asm volatile("" ::: "memory");

        // ---- layer 0 elementwise: update c, write h into A-image ----
        #pragma unroll
        for (int k = 0; k < 26; ++k) {
            const int j = jbase + k;
            if (j < HH) {
                float gi = gates[w][m][j];
                float gf = gates[w][m][HH + j];
                float gg = gates[w][m][2 * HH + j];
                float go = gates[w][m][3 * HH + j];
                float iv = sig_f(gi), fv = sig_f(gf);
                float gv = tanh_f(gg), ov = sig_f(go);
                c[k] = fv * c[k] + iv * gv;
                float hn = ov * tanh_f(c[k]);
                int kk = j & 31, ch = j >> 5;
                int b2 = (kk & 15) >> 3;
                int slot = (kk & 7) + ((kk >> 4) << 3);
                hA[w][ch][m + (b2 << 4)][slot] = (_Float16)hn;
            }
        }
        asm volatile("" ::: "memory");

        // ---- layer 1: gates = [h,1,x] @ [W1; b1; 0]^T ----
        a0 = *(const v16h*)&hA[w][0][lane][0];
        a1 = *(const v16h*)&hA[w][1][lane][0];
        for (int nt = 0; nt < NT; ++nt) {
            const int col = nt * 16 + lo;
            v8f acc = {};
            v16h b0v = *(const v16h*)&sB[1][nt][0][lane][0];
            v16h b1v = *(const v16h*)&sB[1][nt][1][lane][0];
            acc = __builtin_amdgcn_wmma_f32_16x16x32_f16(false, a0, false, b0v,
                                                         (short)0, acc, false, false);
            acc = __builtin_amdgcn_wmma_f32_16x16x32_f16(false, a1, false, b1v,
                                                         (short)0, acc, false, false);
            #pragma unroll
            for (int r = 0; r < 8; ++r) gates[w][hi * 8 + r][col] = acc[r];
        }
        asm volatile("" ::: "memory");

        // ---- layer 1 elementwise + output projection partials ----
        float op = 0.0f;
        #pragma unroll
        for (int k = 0; k < 26; ++k) {
            const int j = jbase + k;
            if (j < HH) {
                float gi = gates[w][m][j];
                float gf = gates[w][m][HH + j];
                float gg = gates[w][m][2 * HH + j];
                float go = gates[w][m][3 * HH + j];
                float iv = sig_f(gi), fv = sig_f(gf);
                float gv = tanh_f(gg), ov = sig_f(go);
                c[k] = fv * c[k] + iv * gv;
                float hn = ov * tanh_f(c[k]);
                int kk = j & 31, ch = j >> 5;
                int b2 = (kk & 15) >> 3;
                int slot = (kk & 7) + ((kk >> 4) << 3);
                hA[w][ch][m + (b2 << 4)][slot] = (_Float16)hn;
                op += hn * sWl[j];
            }
        }
        obuf[w][lane] = op;
        asm volatile("" ::: "memory");
        if (lane < 16) {
            float ov = obuf[w][lane] + obuf[w][lane + 16] + bl;
            out[(size_t)(rowBase + lane) * TLEN + t] = ov;
        }
        xv = xnext;
    }
}

extern "C" void kernel_launch(void* const* d_in, const int* in_sizes, int n_in,
                              void* d_out, int out_size, void* d_ws, size_t ws_size,
                              hipStream_t stream) {
    const float* input = (const float*)d_in[0];
    const float* W_ih0 = (const float*)d_in[1];
    const float* W_hh0 = (const float*)d_in[2];
    const float* b_ih0 = (const float*)d_in[3];
    const float* b_hh0 = (const float*)d_in[4];
    const float* W_ih1 = (const float*)d_in[5];
    const float* W_hh1 = (const float*)d_in[6];
    const float* b_ih1 = (const float*)d_in[7];
    const float* b_hh1 = (const float*)d_in[8];
    const float* Wl    = (const float*)d_in[9];
    const float* bl    = (const float*)d_in[10];
    (void)n_in; (void)out_size; (void)ws_size;

    const int B = in_sizes[0] / TLEN;

    _Float16* Bimg = (_Float16*)d_ws;
    float* fbuf = (float*)((char*)d_ws + (size_t)(2 * NT * 2 * 32 * 16) * 2);

    lstm_prep_kernel<<<1, 256, 0, stream>>>(W_ih0, W_hh0, b_ih0, b_hh0,
                                            W_ih1, W_hh1, b_ih1, b_hh1,
                                            Wl, Bimg, fbuf);

    const int blocks = B / (WPB * RPW);
    lstm_scan_kernel<<<blocks, WPB * 32, 0, stream>>>(input, Bimg, fbuf, bl,
                                                      (float*)d_out, B);
}
